// FFNWithScales_53283364274716
// MI455X (gfx1250) — compile-verified, matrix operand order checked by reading, not run on hardware
//
#include <hip/hip_runtime.h>
#include <math.h>

// ---------------------------------------------------------------------------
// MI455X (gfx1250) SwiGLU FFN with low-rank dequant.
// Memory-bound: ~201MB fp32 weights @ 23.3 TB/s -> ~9us floor. All matmul on
// v_wmma_f32_16x16x32_bf16 (RANK=32 == WMMA K, so scale A@B is ONE wmma/tile).
// Weights dequantized once into bf16 A-fragment order; GEMMs stage fragments
// into double-buffered LDS with async global->LDS copies (ASYNCcnt) when the
// toolchain exposes the gfx1250 builtins, else a synchronous LDS fallback.
// ---------------------------------------------------------------------------

typedef __attribute__((ext_vector_type(16))) __bf16 v16bf;
typedef __attribute__((ext_vector_type(8)))  float  v8f;
typedef int v4i __attribute__((vector_size(16)));   // matches builtin signature

#define D_MODEL 2048
#define D_FF    8192
#define RANK    32
#define SEQ     512

#if defined(__HIP_DEVICE_COMPILE__) && \
    __has_builtin(__builtin_amdgcn_global_load_async_to_lds_b128) && \
    __has_builtin(__builtin_amdgcn_s_wait_asynccnt)
#define HAVE_ASYNC 1
#else
#define HAVE_ASYNC 0
#endif

#if HAVE_ASYNC
#define WAIT_ASYNC0() __builtin_amdgcn_s_wait_asynccnt(0)
#else
#define WAIT_ASYNC0()
#endif

// Copy 16B global -> LDS. Async (ASYNCcnt-tracked) when available.
// Builtin signature (from probe): (v4i AS1* gsrc, v4i AS3* lds, imm offset, imm cpol)
__device__ __forceinline__ void copy16_g2l(void* lds_dst, const void* gsrc) {
#if HAVE_ASYNC
    __builtin_amdgcn_global_load_async_to_lds_b128(
        (__attribute__((address_space(1))) v4i*)(gsrc),
        (__attribute__((address_space(3))) v4i*)(lds_dst),
        0, 0);
#else
    *(uint4*)lds_dst = *(const uint4*)gsrc;
#endif
}

// 16-bit A/B fragment K-index pattern (wave32), per CDNA5 ISA 7.12.2:
// lane l (h = l>>4) element e: e in [0,8): K = 8h+e ; e in [8,16): K = 16+8h+(e-8)
__device__ __forceinline__ int kpat(int h, int e) {
    return (e < 8) ? (8 * h + e) : (16 + 8 * h + (e - 8));
}

__device__ __forceinline__ v8f vzero8() {
    v8f z = {0.f, 0.f, 0.f, 0.f, 0.f, 0.f, 0.f, 0.f};
    return z;
}

// ---------------------------------------------------------------------------
// Dequant + pack: w_eff[o,c] = W[o,c] * (A[o,:]@B[:,c]), bf16, stored in
// A-fragment order: pack[(o_tile*nkt + c_panel)*32 + lane] = 16 bf16 (32B).
// Scale tile = single v_wmma_f32_16x16x32_bf16 (K = RANK = 32).
// One wave handles one 16(M) x 32(K) panel (two 16x16 scale tiles).
// ---------------------------------------------------------------------------
__global__ void __launch_bounds__(256)
dequant_pack_kernel(const float* __restrict__ W,
                    const float* __restrict__ SA,
                    const float* __restrict__ SB,
                    v16bf* __restrict__ pack,
                    int C, int nkt) {
    __shared__ __align__(16) __bf16 sbuf[8][16][40];  // per-wave 16x32 panel (+pad)
    const int lane = threadIdx.x & 31;
    const int w    = threadIdx.x >> 5;
    const int h    = lane >> 4;
    const int r16  = lane & 15;
    const int otile = blockIdx.y;
    const int panel = blockIdx.x * 8 + w;
    const int o0 = otile * 16;
    const int c0 = panel * 32;

    // A fragment of the scale GEMM: 2x8 contiguous fp32 -> 4x float4 (b128 loads)
    const float* arow = &SA[(o0 + r16) * RANK];
    float4 a01 = *(const float4*)(arow + 8 * h);
    float4 a23 = *(const float4*)(arow + 8 * h + 4);
    float4 a45 = *(const float4*)(arow + 16 + 8 * h);
    float4 a67 = *(const float4*)(arow + 16 + 8 * h + 4);
    float af[16] = {a01.x, a01.y, a01.z, a01.w, a23.x, a23.y, a23.z, a23.w,
                    a45.x, a45.y, a45.z, a45.w, a67.x, a67.y, a67.z, a67.w};
    v16bf afrag;
#pragma unroll
    for (int e = 0; e < 16; ++e) afrag[e] = (__bf16)af[e];

#pragma unroll
    for (int t = 0; t < 2; ++t) {
        const int col = c0 + t * 16 + r16;
        v16bf bfrag;
#pragma unroll
        for (int e = 0; e < 16; ++e)
            bfrag[e] = (__bf16)SB[kpat(h, e) * C + col];
        v8f acc = vzero8();
        acc = __builtin_amdgcn_wmma_f32_16x16x32_bf16(false, afrag, false, bfrag,
                                                      (short)0, acc, false, false);
        // D layout: VGPR i, lane l -> (M = i + 8h, N = l%16)
#pragma unroll
        for (int i = 0; i < 8; ++i) {
            float wv = W[(long)(o0 + i + 8 * h) * C + col];
            sbuf[w][i + 8 * h][t * 16 + r16] = (__bf16)(wv * acc[i]);
        }
    }
    // Re-read in A-fragment order (wave-private LDS; compiler inserts dscnt wait)
    v16bf out;
#pragma unroll
    for (int e = 0; e < 16; ++e)
        out[e] = sbuf[w][r16][kpat(h, e)];
    pack[((long)otile * nkt + panel) * 32 + lane] = out;
}

// ---------------------------------------------------------------------------
// Pack x [D_MODEL, SEQ] fp32 -> bf16 B-fragment order:
// xpack[(kt*(SEQ/16) + nt)*32 + lane] = 16 bf16. One wave per (kt, nt).
// ---------------------------------------------------------------------------
__global__ void __launch_bounds__(256)
xpack_kernel(const float* __restrict__ x, v16bf* __restrict__ xp) {
    const int lane = threadIdx.x & 31;
    const int w    = threadIdx.x >> 5;
    const int h    = lane >> 4;
    const int r16  = lane & 15;
    const int gw = blockIdx.x * 8 + w;
    const int NNT = SEQ / 16;                  // 32
    const int nt = gw % NNT;
    const int kt = gw / NNT;                   // 0..63
    v16bf v;
#pragma unroll
    for (int e = 0; e < 16; ++e)
        v[e] = (__bf16)x[(long)(kt * 32 + kpat(h, e)) * SEQ + nt * 16 + r16];
    xp[((long)kt * NNT + nt) * 32 + lane] = v;
}

// ---------------------------------------------------------------------------
// GEMM1 fused gate+up+SwiGLU: mlp = silu(Wg@x) * (Wu@x).
// Block 256 thr = 8 waves (2x4); block tile 64(M) x 256(N); wave tile 32x64.
// Fragments staged via double-buffered LDS (async global->LDS).
// Entries per K-step: 4 gate A + 4 up A + 16 x B = 24 x 1KB.
// Output written directly in GEMM2 B-fragment layout (D-layout == B-layout
// per lane, no cross-lane movement).
// ---------------------------------------------------------------------------
__global__ void __launch_bounds__(256)
gemm1_swiglu_kernel(const char* __restrict__ gpack,
                    const char* __restrict__ upack,
                    const char* __restrict__ xpack,
                    v16bf* __restrict__ mlppack) {
    constexpr int ENT = 24;                    // 1KB entries per K-step
    constexpr int STEP = ENT * 1024;           // 24KB
    __shared__ __align__(32) char smem[2 * STEP];
    const int tid  = threadIdx.x;
    const int lane = tid & 31;
    const int w    = tid >> 5;
    const int wm = w >> 2, wn = w & 3;
    const int bx = blockIdx.x, by = blockIdx.y;
    const int n0 = by * 256 + wn * 64;
    const int NKT = D_MODEL / 32;              // 64
    const int NNT = SEQ / 16;                  // 32

    auto stage = [&](int kt, char* buf) {
#pragma unroll
        for (int j = 0; j < 6; ++j) {          // 24*64/256 = 6 chunks/thread
            const int c = j * 256 + tid;
            const int e = c >> 6;
            const int off = (c & 63) * 16;
            const char* src;
            if (e < 4)
                src = gpack + ((size_t)(bx * 4 + e) * NKT + kt) * 1024 + off;
            else if (e < 8)
                src = upack + ((size_t)(bx * 4 + (e - 4)) * NKT + kt) * 1024 + off;
            else
                src = xpack + ((size_t)kt * NNT + by * 16 + (e - 8)) * 1024 + off;
            copy16_g2l(buf + e * 1024 + off, src);
        }
    };

    v8f gacc[2][4], uacc[2][4];
#pragma unroll
    for (int mi = 0; mi < 2; ++mi)
#pragma unroll
        for (int ni = 0; ni < 4; ++ni) { gacc[mi][ni] = vzero8(); uacc[mi][ni] = vzero8(); }

    stage(0, smem);
    for (int kt = 0; kt < NKT; ++kt) {
        char* cur = smem + (kt & 1) * STEP;
        WAIT_ASYNC0();
        __syncthreads();
        if (kt + 1 < NKT) stage(kt + 1, smem + ((kt + 1) & 1) * STEP);

        v16bf ag[2], au[2], bxf[4];
#pragma unroll
        for (int mi = 0; mi < 2; ++mi) {
            ag[mi] = *(const v16bf*)(cur + (wm * 2 + mi) * 1024 + lane * 32);
            au[mi] = *(const v16bf*)(cur + (4 + wm * 2 + mi) * 1024 + lane * 32);
        }
#pragma unroll
        for (int ni = 0; ni < 4; ++ni)
            bxf[ni] = *(const v16bf*)(cur + (8 + wn * 4 + ni) * 1024 + lane * 32);
#pragma unroll
        for (int mi = 0; mi < 2; ++mi)
#pragma unroll
            for (int ni = 0; ni < 4; ++ni) {
                gacc[mi][ni] = __builtin_amdgcn_wmma_f32_16x16x32_bf16(
                    false, ag[mi], false, bxf[ni], (short)0, gacc[mi][ni], false, false);
                uacc[mi][ni] = __builtin_amdgcn_wmma_f32_16x16x32_bf16(
                    false, au[mi], false, bxf[ni], (short)0, uacc[mi][ni], false, false);
            }
    }

    // Epilogue: silu(g)*u, emit directly in GEMM2 B-fragment layout.
    const int m0 = bx * 64 + wm * 32;
    const int kt2 = m0 / 32;                   // 32-row K panel of GEMM2
#pragma unroll
    for (int ni = 0; ni < 4; ++ni) {
        v16bf o;
#pragma unroll
        for (int e = 0; e < 16; ++e) {
            const int mi = e >> 3, i = e & 7;
            const float g = gacc[mi][ni][i];
            const float u = uacc[mi][ni][i];
            const float s = g / (1.0f + __expf(-g));
            o[e] = (__bf16)(s * u);
        }
        mlppack[((long)kt2 * NNT + (n0 / 16 + ni)) * 32 + lane] = o;
    }
}

// ---------------------------------------------------------------------------
// GEMM2: out[2048,512] fp32 = down_eff @ mlp. Same tiling; 4 A + 16 B entries.
// ---------------------------------------------------------------------------
__global__ void __launch_bounds__(256)
gemm2_kernel(const char* __restrict__ dpack,
             const char* __restrict__ mlppack,
             float* __restrict__ out) {
    constexpr int ENT = 20;
    constexpr int STEP = ENT * 1024;           // 20KB
    __shared__ __align__(32) char smem[2 * STEP];
    const int tid  = threadIdx.x;
    const int lane = tid & 31;
    const int w    = tid >> 5;
    const int wm = w >> 2, wn = w & 3;
    const int bx = blockIdx.x, by = blockIdx.y;
    const int m0 = bx * 64 + wm * 32;
    const int n0 = by * 256 + wn * 64;
    const int NKT = D_FF / 32;                 // 256
    const int NNT = SEQ / 16;                  // 32
    const int h = lane >> 4, r16 = lane & 15;

    auto stage = [&](int kt, char* buf) {
#pragma unroll
        for (int j = 0; j < 5; ++j) {          // 20*64/256 = 5 chunks/thread
            const int c = j * 256 + tid;
            const int e = c >> 6;
            const int off = (c & 63) * 16;
            const char* src;
            if (e < 4)
                src = dpack + ((size_t)(bx * 4 + e) * NKT + kt) * 1024 + off;
            else
                src = mlppack + ((size_t)kt * NNT + by * 16 + (e - 4)) * 1024 + off;
            copy16_g2l(buf + e * 1024 + off, src);
        }
    };

    v8f acc[2][4];
#pragma unroll
    for (int mi = 0; mi < 2; ++mi)
#pragma unroll
        for (int ni = 0; ni < 4; ++ni) acc[mi][ni] = vzero8();

    stage(0, smem);
    for (int kt = 0; kt < NKT; ++kt) {
        char* cur = smem + (kt & 1) * STEP;
        WAIT_ASYNC0();
        __syncthreads();
        if (kt + 1 < NKT) stage(kt + 1, smem + ((kt + 1) & 1) * STEP);

        v16bf a[2], b[4];
#pragma unroll
        for (int mi = 0; mi < 2; ++mi)
            a[mi] = *(const v16bf*)(cur + (wm * 2 + mi) * 1024 + lane * 32);
#pragma unroll
        for (int ni = 0; ni < 4; ++ni)
            b[ni] = *(const v16bf*)(cur + (4 + wn * 4 + ni) * 1024 + lane * 32);
#pragma unroll
        for (int mi = 0; mi < 2; ++mi)
#pragma unroll
            for (int ni = 0; ni < 4; ++ni)
                acc[mi][ni] = __builtin_amdgcn_wmma_f32_16x16x32_bf16(
                    false, a[mi], false, b[ni], (short)0, acc[mi][ni], false, false);
    }

#pragma unroll
    for (int mi = 0; mi < 2; ++mi)
#pragma unroll
        for (int ni = 0; ni < 4; ++ni)
#pragma unroll
            for (int i = 0; i < 8; ++i)
                out[(long)(m0 + mi * 16 + i + 8 * h) * SEQ + n0 + ni * 16 + r16] =
                    acc[mi][ni][i];
}

// ---------------------------------------------------------------------------
extern "C" void kernel_launch(void* const* d_in, const int* in_sizes, int n_in,
                              void* d_out, int out_size, void* d_ws, size_t ws_size,
                              hipStream_t stream) {
    const float* x        = (const float*)d_in[0];
    const float* gate_w   = (const float*)d_in[1];
    const float* gate_sa  = (const float*)d_in[2];
    const float* gate_sb  = (const float*)d_in[3];
    const float* up_w     = (const float*)d_in[4];
    const float* up_sa    = (const float*)d_in[5];
    const float* up_sb    = (const float*)d_in[6];
    const float* down_w   = (const float*)d_in[7];
    const float* down_sa  = (const float*)d_in[8];
    const float* down_sb  = (const float*)d_in[9];
    float* out = (float*)d_out;

    // Workspace layout (bf16): gate_pack 32MB | up_pack 32MB | down_pack 32MB |
    //                          x_pack 2MB | mlp_pack 8MB   (total ~106MB)
    char* ws = (char*)d_ws;
    const size_t GUP = (size_t)D_FF * D_MODEL * sizeof(__bf16);  // 32MB each
    v16bf* gate_pack = (v16bf*)(ws);
    v16bf* up_pack   = (v16bf*)(ws + GUP);
    v16bf* down_pack = (v16bf*)(ws + 2 * GUP);
    v16bf* x_pack    = (v16bf*)(ws + 3 * GUP);
    v16bf* mlp_pack  = (v16bf*)(ws + 3 * GUP + (size_t)D_MODEL * SEQ * sizeof(__bf16));

    dim3 blk(256);

    // Dequantize + pack all three weight matrices (bf16, A-fragment order).
    dequant_pack_kernel<<<dim3((D_MODEL / 32) / 8, D_FF / 16), blk, 0, stream>>>(
        gate_w, gate_sa, gate_sb, gate_pack, D_MODEL, D_MODEL / 32);
    dequant_pack_kernel<<<dim3((D_MODEL / 32) / 8, D_FF / 16), blk, 0, stream>>>(
        up_w, up_sa, up_sb, up_pack, D_MODEL, D_MODEL / 32);
    dequant_pack_kernel<<<dim3((D_FF / 32) / 8, D_MODEL / 16), blk, 0, stream>>>(
        down_w, down_sa, down_sb, down_pack, D_FF, D_FF / 32);

    // Pack activations.
    xpack_kernel<<<dim3((D_MODEL / 32) * (SEQ / 16) / 8), blk, 0, stream>>>(x, x_pack);

    // Fused gate/up GEMM + SwiGLU.
    gemm1_swiglu_kernel<<<dim3(D_FF / 64, SEQ / 256), blk, 0, stream>>>(
        (const char*)gate_pack, (const char*)up_pack, (const char*)x_pack, mlp_pack);

    // Down projection.
    gemm2_kernel<<<dim3(D_MODEL / 64, SEQ / 256), blk, 0, stream>>>(
        (const char*)down_pack, (const char*)mlp_pack, out);
}
